// PVRCNNPlusPlusVoxelSetAbstraction_730144440352
// MI455X (gfx1250) — matrix operand order repaired and checked by reading, not compiled, not run on hardware
//
#include <hip/hip_runtime.h>
#include <hip/hip_bf16.h>

// ---------------- problem constants (from reference setup_inputs) ------------
#define B_   2
#define N_   16384      // raw points per batch
#define M_   8192       // conv3 points per batch
#define R_   128        // rois
#define KP_  2048       // keypoints
#define HW_  188        // BEV H=W
#define CBEV_ 256       // BEV channels
#define CF_  352        // concat feature width (256+32+64)
#define NO_  128        // output channels
#define NKT_ 11         // CF_/32 K-tiles in fuse GEMM

// ---------------- CDNA5 WMMA types ------------------------------------------
typedef __attribute__((ext_vector_type(16))) __bf16        v16bf;
typedef __attribute__((ext_vector_type(8)))  float         v8f;
typedef __attribute__((ext_vector_type(4)))  unsigned int  u32x4;

union V16 { u32x4 q[2]; unsigned short u[16]; v16bf v; };

static __device__ __forceinline__ unsigned short f2bf(float f) {
  unsigned int x = __float_as_uint(f);
  x += 0x7FFFu + ((x >> 16) & 1u);          // round-to-nearest-even
  return (unsigned short)(x >> 16);
}

// position of (row,col) element inside the fuse-GEMM A-operand swizzled buffer
// layout: [(mt*11 + kt)][lane(0..31)][e(0..15)]  (bf16)
static __device__ __forceinline__ size_t fuseA_idx(int row, int col) {
  int mt = row >> 4, m2 = row & 15;
  int kt = col >> 5, kk = col & 31;
  int kh = (kk >> 3) & 1;                       // K-half of lane
  int e  = (kk & 7) | (((kk >> 4) & 1) << 3);   // element within fragment
  int lane = m2 + kh * 16;
  return (((size_t)mt * NKT_ + kt) * 32 + lane) * 16 + e;
}

// =============================================================================
// Kernel 0: swizzle an f32 weight matrix (K x NC, row-major) into packed-bf16
// WMMA B-operand order: out[((kt*NT + n)*32 + lane)*16 + e]
// =============================================================================
__global__ void k_swzW(const float* __restrict__ W, int K, int NC,
                       unsigned short* __restrict__ out) {
  int id = blockIdx.x * blockDim.x + threadIdx.x;
  if (id >= K * NC) return;
  int NT = NC >> 4;
  int e  = id & 15;
  int ln = (id >> 4) & 31;
  int rest = id >> 9;
  int n  = rest % NT;
  int kt = rest / NT;
  int kh = ln >> 4;
  int col  = n * 16 + (ln & 15);
  int krow = kt * 32 + kh * 8 + (e & 7) + ((e >> 3) << 4);
  out[id] = f2bf(W[(size_t)krow * NC + col]);
}

// =============================================================================
// Kernel 1: ROI validity -> FPS seed distances (+1e10 valid / -1e10 invalid)
// =============================================================================
__global__ void k_valid(const float* __restrict__ pts,
                        const float* __restrict__ boxes,
                        float* __restrict__ d0) {
  int gid = blockIdx.x * blockDim.x + threadIdx.x;      // over B*N
  if (gid >= B_ * N_) return;
  int b = gid / N_;
  const float* p = pts + (size_t)gid * 5;
  float x = p[0], y = p[1], z = p[2];
  const float* bx = boxes + (size_t)b * R_ * 7;
  float best = 3.4e38f; int bi = 0;
  for (int r = 0; r < R_; ++r) {
    float dx = x - bx[r*7+0], dy = y - bx[r*7+1], dz = z - bx[r*7+2];
    float d = sqrtf(dx*dx + dy*dy + dz*dz);
    if (d < best) { best = d; bi = r; }                 // first-min tiebreak
  }
  float hx = bx[bi*7+3]*0.5f, hy = bx[bi*7+4]*0.5f, hz = bx[bi*7+5]*0.5f;
  float rmax = sqrtf(hx*hx + hy*hy + hz*hz);
  d0[gid] = (best < rmax + 2.4f) ? 1e10f : -1e10f;
}

// =============================================================================
// Kernel 2: Farthest point sampling. One workgroup per batch, fully register-
// resident xyz+dist (16 pts/thread), wave32 shuffle butterfly + LDS top-level.
// =============================================================================
__global__ __launch_bounds__(1024)
void k_fps(const float* __restrict__ pts,
           const float* __restrict__ d0,
           float* __restrict__ kp /* B*KP*3 */) {
  int b = blockIdx.x;
  int t = threadIdx.x;
  int lane = t & 31, w = t >> 5;
  const float* P = pts + (size_t)b * N_ * 5;
  float px[16], py[16], pz[16], dd[16];
#pragma unroll
  for (int j = 0; j < 16; ++j) {
    int n = t * 16 + j;
    px[j] = P[n*5+0]; py[j] = P[n*5+1]; pz[j] = P[n*5+2];
    dd[j] = d0[b * N_ + n];
  }
  __shared__ float s_v[32], s_x[32], s_y[32], s_z[32];
  __shared__ int   s_i[32];
  __shared__ float s_cx, s_cy, s_cz;

  for (int k = 0; k < KP_; ++k) {
    float bv = -3.4e38f, bxx = 0.f, byy = 0.f, bzz = 0.f;
    int bi = 0x7fffffff;
#pragma unroll
    for (int j = 0; j < 16; ++j)
      if (dd[j] > bv) { bv = dd[j]; bi = t*16 + j; bxx = px[j]; byy = py[j]; bzz = pz[j]; }
#pragma unroll
    for (int off = 16; off >= 1; off >>= 1) {
      float ov = __shfl_xor(bv, off, 32);
      int   oi = __shfl_xor(bi, off, 32);
      float ox = __shfl_xor(bxx, off, 32);
      float oy = __shfl_xor(byy, off, 32);
      float oz = __shfl_xor(bzz, off, 32);
      if (ov > bv || (ov == bv && oi < bi)) { bv = ov; bi = oi; bxx = ox; byy = oy; bzz = oz; }
    }
    if (lane == 0) { s_v[w] = bv; s_i[w] = bi; s_x[w] = bxx; s_y[w] = byy; s_z[w] = bzz; }
    __syncthreads();
    if (t == 0) {
      float v = s_v[0]; int i = s_i[0];
      float cx = s_x[0], cy = s_y[0], cz = s_z[0];
      for (int q = 1; q < 32; ++q)
        if (s_v[q] > v || (s_v[q] == v && s_i[q] < i)) {
          v = s_v[q]; i = s_i[q]; cx = s_x[q]; cy = s_y[q]; cz = s_z[q];
        }
      s_cx = cx; s_cy = cy; s_cz = cz;
      float* o = kp + ((size_t)b * KP_ + k) * 3;
      o[0] = cx; o[1] = cy; o[2] = cz;
    }
    __syncthreads();
    float cx = s_cx, cy = s_cy, cz = s_cz;
#pragma unroll
    for (int j = 0; j < 16; ++j) {
      float dx = px[j]-cx, dy = py[j]-cy, dz = pz[j]-cz;
      dd[j] = fminf(dd[j], dx*dx + dy*dy + dz*dz);
    }
  }
}

// =============================================================================
// Kernel 3: f_raw = relu(pfeat @ W_raw + b) -> bf16, written directly in the
// aggregation-GEMM B-operand swizzled layout.
// =============================================================================
__global__ void k_fraw(const float* __restrict__ pts,
                       const float* __restrict__ W, const float* __restrict__ bias,
                       unsigned short* __restrict__ out) {
  int gid = blockIdx.x * blockDim.x + threadIdx.x;      // B*N
  if (gid >= B_ * N_) return;
  int b = gid / N_, p = gid % N_;
  float f0 = pts[(size_t)gid*5 + 3], f1 = pts[(size_t)gid*5 + 4];
  int ktile = p >> 5, kin = p & 31, kh = kin >> 4, e = kin & 15;
  unsigned short* o = out + (size_t)b * N_ * 32;
#pragma unroll
  for (int j = 0; j < 32; ++j) {
    float v = fmaxf(f0 * W[j] + f1 * W[32 + j] + bias[j], 0.f);
    int n = j >> 4, ln = (j & 15) + kh * 16;
    o[(((size_t)ktile * 2 + n) * 32 + ln) * 16 + e] = f2bf(v);
  }
}

// =============================================================================
// Kernel 4: f_c3 = relu(cfeat(8192x64) @ W_c3(64x64) + b) via bf16 WMMA.
// B operand: pre-swizzled bf16 (2x b128 loads). Output written in B-swizzled
// layout for the aggregation GEMM.
// =============================================================================
__global__ __launch_bounds__(32)
void k_fc3(const float* __restrict__ conv3,
           const unsigned short* __restrict__ Wb, const float* __restrict__ bias,
           unsigned short* __restrict__ out) {
  int tile = blockIdx.x;                 // B * M/16
  int b  = tile / (M_/16);
  int rt = tile % (M_/16);
  int lane = threadIdx.x & 31;
  int m = lane & 15, kh = lane >> 4;
  int row = rt*16 + m;
  const float* src = conv3 + ((size_t)b * M_ + row) * 67 + 3;
  v8f acc[4];
#pragma unroll
  for (int n = 0; n < 4; ++n)
#pragma unroll
    for (int e = 0; e < 8; ++e) acc[n][e] = 0.f;

#pragma unroll
  for (int kt = 0; kt < 2; ++kt) {
    V16 a;
#pragma unroll
    for (int e = 0; e < 16; ++e) {
      int kk = (e < 8) ? (kh*8 + e) : (16 + kh*8 + (e - 8));   // 16-bit A layout
      a.u[e] = f2bf(src[kt*32 + kk]);
    }
#pragma unroll
    for (int n = 0; n < 4; ++n) {
      V16 bb;
      const u32x4* pB = (const u32x4*)(Wb + ((size_t)(kt*4 + n)*32 + lane)*16);
      bb.q[0] = pB[0]; bb.q[1] = pB[1];
      acc[n] = __builtin_amdgcn_wmma_f32_16x16x32_bf16(
          false, a.v, false, bb.v, (short)0, acc[n], false, false);
    }
  }
  // scatter into agg-B swizzled layout (width 64 -> 4 N-tiles)
  int ktile = rt >> 1, kh2 = rt & 1;
  unsigned short* o = out + (size_t)b * M_ * 64;
#pragma unroll
  for (int n = 0; n < 4; ++n) {
    int col = n*16 + m;
    float bi = bias[col];
#pragma unroll
    for (int r = 0; r < 8; ++r) {
      int e2   = r + kh*8;                 // point-in-halftile
      int ln2  = m + kh2*16;               // (feature%16, K-half)
      float v = fmaxf(acc[n][r] + bi, 0.f);
      o[(((size_t)ktile * 4 + n) * 32 + ln2) * 16 + e2] = f2bf(v);
    }
  }
}

// =============================================================================
// Kernel 5: BEV bilinear gather -> write cols 0..255 of fuse-A swizzled buffer
// =============================================================================
__global__ void k_bilinear(const float* __restrict__ sf,
                           const float* __restrict__ kp,
                           unsigned short* __restrict__ fA) {
  int bk = blockIdx.x;                   // B*KP (= output row)
  int b = bk / KP_;
  int c = threadIdx.x;                   // 256 channels
  const float* K = kp + (size_t)bk * 3;
  float xi = (K[0] + 75.2f) * 1.25f;     // (x - PCR0)/VSZ0/8
  float yi = (K[1] + 75.2f) * 1.25f;
  int x0 = (int)floorf(xi); x0 = x0 < 0 ? 0 : (x0 > HW_-1 ? HW_-1 : x0);
  int x1 = x0 + 1 > HW_-1 ? HW_-1 : x0 + 1;
  int y0 = (int)floorf(yi); y0 = y0 < 0 ? 0 : (y0 > HW_-1 ? HW_-1 : y0);
  int y1 = y0 + 1 > HW_-1 ? HW_-1 : y0 + 1;
  float xf0 = (float)x0, xf1 = (float)x1, yf0 = (float)y0, yf1 = (float)y1;
  float wa = (xf1 - xi) * (yf1 - yi);
  float wb = (xf1 - xi) * (yi - yf0);
  float wc = (xi - xf0) * (yf1 - yi);
  float wd = (xi - xf0) * (yi - yf0);
  const float* im = sf + ((size_t)b * CBEV_ + c) * (HW_ * HW_);
  float Ia = im[y0*HW_ + x0], Ib = im[y1*HW_ + x0];
  float Ic = im[y0*HW_ + x1], Id = im[y1*HW_ + x1];
  float v = Ia*wa + Ib*wb + Ic*wc + Id*wd;
  fA[fuseA_idx(bk, c)] = f2bf(v);
}

// =============================================================================
// Kernel 6: fused radius-mask aggregation:  out = (m @ f) / max(rowsum(m),1)
// Mask built in-register directly in WMMA A layout; f read as pre-swizzled
// bf16 B fragments via contiguous b128 loads. Result written into fuse-A
// swizzled buffer. FW = feature N-tiles (2 for raw, 4 for conv3).
// =============================================================================
template <int FW>
__global__ __launch_bounds__(128)
void k_agg(const float* __restrict__ sxyz, int stride, int npts,
           const unsigned short* __restrict__ f,   // B-swizzled bf16
           const float* __restrict__ kp,
           float r2, int outoff,
           unsigned short* __restrict__ fA) {
  int b = blockIdx.y;
  int lane = threadIdx.x & 31, w = threadIdx.x >> 5;
  int m = lane & 15, kh = lane >> 4;
  int kprow = blockIdx.x * 64 + w * 16 + m;
  const float* kpp = kp + ((size_t)b * KP_ + kprow) * 3;
  float kx = kpp[0], ky = kpp[1], kz = kpp[2];

  __shared__ float sx[32], sy[32], sz[32];

  v8f acc[FW];
#pragma unroll
  for (int n = 0; n < FW; ++n)
#pragma unroll
    for (int e = 0; e < 8; ++e) acc[n][e] = 0.f;
  float cnt = 0.f;

  const float* X = sxyz + (size_t)b * npts * stride;
  const unsigned short* F = f + (size_t)b * npts * (FW * 16);
  const int nkt = npts / 32;

  for (int kt = 0; kt < nkt; ++kt) {
    if (threadIdx.x < 32) {
      const float* q = X + (size_t)(kt*32 + threadIdx.x) * stride;
      sx[threadIdx.x] = q[0]; sy[threadIdx.x] = q[1]; sz[threadIdx.x] = q[2];
    }
    __syncthreads();
    if (kt + 1 < nkt)
      __builtin_prefetch(F + ((size_t)(kt + 1) * FW) * 32 * 16, 0, 1);  // global_prefetch_b8

    V16 a;
#pragma unroll
    for (int e = 0; e < 16; ++e) {
      int kk = kh*16 + e;
      float dx = kx - sx[kk], dy = ky - sy[kk], dz = kz - sz[kk];
      bool in = (dx*dx + dy*dy + dz*dz) < r2;
      a.u[e] = in ? (unsigned short)0x3F80 : (unsigned short)0;     // bf16 1.0 / 0.0
      cnt += in ? 1.f : 0.f;
    }
#pragma unroll
    for (int n = 0; n < FW; ++n) {
      V16 bb;
      const u32x4* pB = (const u32x4*)(F + ((size_t)(kt*FW + n)*32 + lane)*16);
      bb.q[0] = pB[0]; bb.q[1] = pB[1];
      acc[n] = __builtin_amdgcn_wmma_f32_16x16x32_bf16(
          false, a.v, false, bb.v, (short)0, acc[n], false, false);
    }
    __syncthreads();
  }

  cnt += __shfl_xor(cnt, 16, 32);        // full row count (lanes q and q+16)
#pragma unroll
  for (int n = 0; n < FW; ++n) {
#pragma unroll
    for (int r = 0; r < 8; ++r) {
      int orow = r + kh*8;
      float cr = fmaxf(__shfl(cnt, orow, 32), 1.f);
      int row = b * KP_ + blockIdx.x * 64 + w * 16 + orow;
      int col = outoff + n*16 + m;
      fA[fuseA_idx(row, col)] = f2bf(acc[n][r] / cr);
    }
  }
}

// =============================================================================
// Kernel 7: fuse GEMM (4096x352 @ 352x128) + batchnorm + relu via bf16 WMMA.
// A and B both pre-swizzled bf16 -> pure b128 loads + WMMA in the loop.
// =============================================================================
__global__ __launch_bounds__(32)
void k_fuse(const unsigned short* __restrict__ fA,
            const unsigned short* __restrict__ Wb,
            const float* __restrict__ gamma, const float* __restrict__ beta,
            const float* __restrict__ mean,  const float* __restrict__ var,
            float* __restrict__ out) {
  int mt = blockIdx.x;                   // (B*KP)/16
  int lane = threadIdx.x & 31;
  int m = lane & 15, kh = lane >> 4;
  const unsigned short* A = fA + (size_t)mt * NKT_ * 512;
  v8f acc[8];
#pragma unroll
  for (int n = 0; n < 8; ++n)
#pragma unroll
    for (int e = 0; e < 8; ++e) acc[n][e] = 0.f;

#pragma unroll
  for (int kt = 0; kt < NKT_; ++kt) {    // 11 K-steps
    V16 a;
    const u32x4* pA = (const u32x4*)(A + ((size_t)kt*32 + lane)*16);
    a.q[0] = pA[0]; a.q[1] = pA[1];
#pragma unroll
    for (int n = 0; n < 8; ++n) {
      V16 bb;
      const u32x4* pB = (const u32x4*)(Wb + ((size_t)(kt*8 + n)*32 + lane)*16);
      bb.q[0] = pB[0]; bb.q[1] = pB[1];
      acc[n] = __builtin_amdgcn_wmma_f32_16x16x32_bf16(
          false, a.v, false, bb.v, (short)0, acc[n], false, false);
    }
  }
#pragma unroll
  for (int n = 0; n < 8; ++n) {
    int col = n*16 + m;
    float sc = gamma[col] * rsqrtf(var[col] + 1e-5f);
    float mu = mean[col], bt = beta[col];
#pragma unroll
    for (int r = 0; r < 8; ++r) {
      int orow = mt*16 + r + kh*8;
      float v = (acc[n][r] - mu) * sc + bt;
      out[(size_t)orow * NO_ + col] = fmaxf(v, 0.f);
    }
  }
}

// =============================================================================
// host-side launch
// =============================================================================
extern "C" void kernel_launch(void* const* d_in, const int* in_sizes, int n_in,
                              void* d_out, int out_size, void* d_ws, size_t ws_size,
                              hipStream_t stream) {
  (void)in_sizes; (void)n_in; (void)out_size; (void)ws_size;
  const float* pts   = (const float*)d_in[0];   // (2,16384,5)
  const float* boxes = (const float*)d_in[1];   // (2,128,7)
  const float* sf    = (const float*)d_in[2];   // (2,256,188,188)
  const float* conv3 = (const float*)d_in[3];   // (2,8192,67)
  const float* Wraw  = (const float*)d_in[4];   // (2,32)
  const float* braw  = (const float*)d_in[5];   // (32)
  const float* Wc3   = (const float*)d_in[6];   // (64,64)
  const float* bc3   = (const float*)d_in[7];   // (64)
  const float* Wfuse = (const float*)d_in[8];   // (352,128)
  const float* gamma = (const float*)d_in[9];
  const float* beta  = (const float*)d_in[10];
  const float* mean  = (const float*)d_in[11];
  const float* var   = (const float*)d_in[12];

  // workspace layout (~7.4 MiB total, all 256B aligned)
  char* ws = (char*)d_ws;
  float*          d0   = (float*)(ws);                      //  524288 B
  unsigned short* fA   = (unsigned short*)(ws + 524288);    // 2883584 B (4096x352 bf16, A-swz)
  unsigned short* fraw = (unsigned short*)(ws + 3407872);   // 2097152 B (B-swz bf16)
  unsigned short* fc3  = (unsigned short*)(ws + 5505024);   // 2097152 B (B-swz bf16)
  unsigned short* Wbf  = (unsigned short*)(ws + 7602176);   //   90112 B (352x128 B-swz)
  unsigned short* Wbc  = (unsigned short*)(ws + 7692288);   //    8192 B (64x64 B-swz)

  float* out = (float*)d_out;                   // (4096,128)
  float* kp  = out + (size_t)B_ * KP_ * NO_;    // (2,2048,3)

  k_swzW    <<<(352*128 + 255)/256, 256, 0, stream>>>(Wfuse, 352, 128, Wbf);
  k_swzW    <<<(64*64   + 255)/256, 256, 0, stream>>>(Wc3,   64,  64,  Wbc);
  k_valid   <<<(B_*N_ + 255)/256, 256, 0, stream>>>(pts, boxes, d0);
  k_fps     <<<B_, 1024, 0, stream>>>(pts, d0, kp);
  k_fraw    <<<(B_*N_ + 255)/256, 256, 0, stream>>>(pts, Wraw, braw, fraw);
  k_fc3     <<<B_*(M_/16), 32, 0, stream>>>(conv3, Wbc, bc3, fc3);
  k_bilinear<<<B_*KP_, CBEV_, 0, stream>>>(sf, kp, fA);
  k_agg<2>  <<<dim3(KP_/64, B_), 128, 0, stream>>>(pts,   5, N_, fraw, kp, 0.64f, 256, fA);
  k_agg<4>  <<<dim3(KP_/64, B_), 128, 0, stream>>>(conv3, 67, M_, fc3,  kp, 2.56f, 288, fA);
  k_fuse    <<<(B_*KP_)/16, 32, 0, stream>>>(fA, Wbf, gamma, beta, mean, var, out);
}
